// UGFR_36180804501795
// MI455X (gfx1250) — compile-verified, matrix operand order checked by reading, not run on hardware
//
#include <hip/hip_runtime.h>
#include <hip/hip_bf16.h>
#include <math.h>

// ---------------------------------------------------------------------------
// Problem constants (from the reference)
// ---------------------------------------------------------------------------
constexpr int BB = 4, HH = 256, WW = 256, HWp = HH * WW;   // 65536 pixels/img
constexpr int C_IN2 = 62, C31 = 31, CM = 3, EMB = 128;
constexpr int RWIN = 2, NOFF = (2 * RWIN + 1) * (2 * RWIN + 1); // 25
constexpr int REG = EMB + EMB + NOFF + 1;                  // 282
constexpr int REGP = 288;                                  // padded to 9*32
constexpr float MAXFLOW = 2.0f;

typedef __attribute__((ext_vector_type(16))) __bf16 v16bf;
typedef __attribute__((ext_vector_type(8)))  float  v8f;

union ABfrag { uint4 u[2]; v16bf v; };

// ---------------------------------------------------------------------------
// Workspace layout (bytes, all 256-aligned)
// ---------------------------------------------------------------------------
constexpr size_t SZ_X31  = (size_t)BB * C31 * HWp * 4;        // 32.5 MB  f32 NCHW
constexpr size_t SZ_XP   = (size_t)BB * CM  * HWp * 4;        // 3.1 MB
constexpr size_t SZ_XA   = SZ_XP;
constexpr size_t SZ_ST   = 256;                               // 12*(sm,ss,rm,rs)
constexpr size_t SZ_SS   = SZ_XP;                             // ssim per-channel
constexpr size_t SZ_DF   = (size_t)BB * HWp * 4;              // difficulty
constexpr size_t SZ_REG  = (size_t)BB * HWp * REGP * 2;       // 151 MB bf16 NHWC
constexpr size_t SZ_H1   = (size_t)BB * HWp * 128  * 2;       // 67 MB  bf16 NHWC
constexpr size_t SZ_H2   = (size_t)BB * HWp * 64   * 4;       // 67 MB  f32  NHWC
constexpr size_t SZ_FL   = (size_t)BB * HWp * 2    * 4;       // flow
constexpr size_t SZ_WP1  = (size_t)8 * 9 * 9 * 32 * 16 * 2;   // 648 KB
constexpr size_t SZ_WP2  = (size_t)4 * 9 * 4 * 32 * 16 * 2;   // 144 KB

constexpr size_t OFF_X31 = 0;
constexpr size_t OFF_XP  = OFF_X31 + SZ_X31;
constexpr size_t OFF_XA  = OFF_XP  + SZ_XP;
constexpr size_t OFF_ST  = OFF_XA  + SZ_XA;
constexpr size_t OFF_SS  = OFF_ST  + SZ_ST;
constexpr size_t OFF_DF  = OFF_SS  + SZ_SS;
constexpr size_t OFF_REG = OFF_DF  + SZ_DF;
constexpr size_t OFF_H1  = OFF_REG + SZ_REG;
constexpr size_t OFF_H2  = OFF_H1  + SZ_H1;
constexpr size_t OFF_FL  = OFF_H2  + SZ_H2;
constexpr size_t OFF_WP1 = OFF_FL  + SZ_FL;
constexpr size_t OFF_WP2 = OFF_WP1 + SZ_WP1;

// ---------------------------------------------------------------------------
// Helpers
// ---------------------------------------------------------------------------
static __device__ __forceinline__ unsigned short f32_to_bf16(float f) {
    unsigned int u = __float_as_uint(f);
    unsigned int r = u + 0x7FFFu + ((u >> 16) & 1u);   // RNE
    return (unsigned short)(r >> 16);
}
static __device__ __forceinline__ float bf16_to_f32(unsigned short h) {
    return __uint_as_float(((unsigned int)h) << 16);
}
static __device__ __forceinline__ float clampf(float v, float lo, float hi) {
    return fminf(fmaxf(v, lo), hi);
}

// ---------------------------------------------------------------------------
// 1) conv_half: 1x1, 62 -> 31, f32 NCHW
// ---------------------------------------------------------------------------
__global__ void k_conv_half(const float* __restrict__ x, const float* __restrict__ w,
                            const float* __restrict__ b, float* __restrict__ x31) {
    int idx = blockIdx.x * blockDim.x + threadIdx.x;
    if (idx >= BB * C31 * HWp) return;
    int p = idx % HWp;
    int c = (idx / HWp) % C31;
    int bb = idx / (C31 * HWp);
    const float* xb = x + (size_t)bb * C_IN2 * HWp + p;
    const float* wr = w + c * C_IN2;
    float s = b[c];
    for (int i = 0; i < C_IN2; ++i) s += xb[(size_t)i * HWp] * wr[i];
    x31[idx] = s;
}

// ---------------------------------------------------------------------------
// 2) SRF projection 31 -> 3, clip
// ---------------------------------------------------------------------------
__global__ void k_project(const float* __restrict__ x31, const float* __restrict__ srf,
                          float* __restrict__ xp) {
    int idx = blockIdx.x * blockDim.x + threadIdx.x;
    if (idx >= BB * CM * HWp) return;
    int p = idx % HWp;
    int m = (idx / HWp) % CM;
    int bb = idx / (CM * HWp);
    const float* xb = x31 + (size_t)bb * C31 * HWp + p;
    float s = 0.f;
    for (int c = 0; c < C31; ++c) s += xb[(size_t)c * HWp] * srf[m * C31 + c];
    xp[idx] = clampf(s, -1.f, 1.f);
}

// ---------------------------------------------------------------------------
// 3) per (b,ch) mean/std (ddof=1) of xp (src) and y (ref)
// ---------------------------------------------------------------------------
__global__ void k_stats(const float* __restrict__ xp, const float* __restrict__ yin,
                        float* __restrict__ stats) {
    int bm = blockIdx.x;                 // 0..11
    const float* s = xp  + (size_t)bm * HWp;
    const float* r = yin + (size_t)bm * HWp;
    float s1 = 0, s2 = 0, r1 = 0, r2 = 0;
    for (int i = threadIdx.x; i < HWp; i += blockDim.x) {
        float a = s[i], b = r[i];
        s1 += a; s2 += a * a; r1 += b; r2 += b * b;
    }
    __shared__ float sh[4][256];
    int t = threadIdx.x;
    sh[0][t] = s1; sh[1][t] = s2; sh[2][t] = r1; sh[3][t] = r2;
    __syncthreads();
    for (int ofs = 128; ofs > 0; ofs >>= 1) {
        if (t < ofs)
            for (int k = 0; k < 4; ++k) sh[k][t] += sh[k][t + ofs];
        __syncthreads();
    }
    if (t == 0) {
        float n = (float)HWp;
        float sm = sh[0][0] / n;
        float sv = (sh[1][0] - n * sm * sm) / (n - 1.f);
        float rm = sh[2][0] / n;
        float rv = (sh[3][0] - n * rm * rm) / (n - 1.f);
        stats[bm * 4 + 0] = sm;
        stats[bm * 4 + 1] = sqrtf(fmaxf(sv, 0.f));
        stats[bm * 4 + 2] = rm;
        stats[bm * 4 + 3] = sqrtf(fmaxf(rv, 0.f));
    }
}

__global__ void k_align(const float* __restrict__ xp, const float* __restrict__ stats,
                        float* __restrict__ xa) {
    int idx = blockIdx.x * blockDim.x + threadIdx.x;
    if (idx >= BB * CM * HWp) return;
    int bm = idx / HWp;
    float sm = stats[bm * 4 + 0], ss = stats[bm * 4 + 1];
    float rm = stats[bm * 4 + 2], rs = stats[bm * 4 + 3];
    xa[idx] = clampf((xp[idx] - sm) / (ss + 1e-6f) * rs + rm, -1.f, 1.f);
}

// ---------------------------------------------------------------------------
// 4) SSIM map (9x9 gaussian, zero-pad) per channel
// ---------------------------------------------------------------------------
__global__ void k_ssim(const float* __restrict__ a, const float* __restrict__ bimg,
                       float* __restrict__ out) {
    int idx = blockIdx.x * blockDim.x + threadIdx.x;
    if (idx >= BB * CM * HWp) return;
    int p = idx % HWp;
    int x = p % WW, y = p / WW;
    int bm = idx / HWp;
    const float* ap = a    + (size_t)bm * HWp;
    const float* bp = bimg + (size_t)bm * HWp;

    float g[9]; float gs = 0.f;
    for (int i = 0; i < 9; ++i) { float c = (float)i - 4.f; g[i] = __expf(-c * c / 4.5f); gs += g[i]; }
    for (int i = 0; i < 9; ++i) g[i] /= gs;

    float mu1 = 0, mu2 = 0, saa = 0, sbb = 0, sab = 0;
    for (int dy = 0; dy < 9; ++dy) {
        int yy = y + dy - 4;
        if (yy < 0 || yy >= HH) continue;
        float wy = g[dy];
        const float* ar = ap + yy * WW;
        const float* br = bp + yy * WW;
        for (int dx = 0; dx < 9; ++dx) {
            int xx = x + dx - 4;
            if (xx < 0 || xx >= WW) continue;
            float w = wy * g[dx];
            float av = ar[xx], bv = br[xx];
            mu1 += w * av; mu2 += w * bv;
            saa += w * av * av; sbb += w * bv * bv; sab += w * av * bv;
        }
    }
    float s1 = saa - mu1 * mu1, s2 = sbb - mu2 * mu2, s12 = sab - mu1 * mu2;
    const float C1 = 1e-4f, C2 = 9e-4f;
    float m = ((2.f * mu1 * mu2 + C1) * (2.f * s12 + C2)) /
              ((mu1 * mu1 + mu2 * mu2 + C1) * (s1 + s2 + C2));
    out[idx] = m;
}

// ---------------------------------------------------------------------------
// 5) difficulty + reg_in tail (ch 281 + zero pad 282..287)
// ---------------------------------------------------------------------------
__global__ void k_diff(const float* __restrict__ ssim3, float* __restrict__ diff,
                       unsigned short* __restrict__ reg) {
    int idx = blockIdx.x * blockDim.x + threadIdx.x;
    if (idx >= BB * HWp) return;
    int p = idx % HWp, bb = idx / HWp;
    float m = (ssim3[((size_t)bb * 3 + 0) * HWp + p] +
               ssim3[((size_t)bb * 3 + 1) * HWp + p] +
               ssim3[((size_t)bb * 3 + 2) * HWp + p]) * (1.f / 3.f);
    float d = clampf((1.f - m) * 0.5f, 0.f, 1.f);
    diff[idx] = d;
    unsigned short* rp = reg + (size_t)idx * REGP;
    rp[281] = f32_to_bf16(d);
    for (int c = 282; c < REGP; ++c) rp[c] = 0;
}

// ---------------------------------------------------------------------------
// 6) q (31->128) and k (3->128) 1x1 convs -> reg_in bf16 channels 0..255
// ---------------------------------------------------------------------------
__global__ void k_qk(const float* __restrict__ x31, const float* __restrict__ yin,
                     const float* __restrict__ qw, const float* __restrict__ qb,
                     const float* __restrict__ kw, const float* __restrict__ kb,
                     unsigned short* __restrict__ reg) {
    int gp = blockIdx.x;                 // 0..B*HW-1
    int bb = gp / HWp, p = gp % HWp;
    __shared__ float xs[C31];
    __shared__ float ys[CM];
    int t = threadIdx.x;                 // 128
    if (t < C31) xs[t] = x31[((size_t)bb * C31 + t) * HWp + p];
    if (t < CM)  ys[t] = yin[((size_t)bb * CM  + t) * HWp + p];
    __syncthreads();
    float q = qb[t];
    for (int c = 0; c < C31; ++c) q += xs[c] * qw[t * C31 + c];
    float kk = kb[t];
    for (int c = 0; c < CM; ++c) kk += ys[c] * kw[t * CM + c];
    unsigned short* rp = reg + (size_t)gp * REGP;
    rp[t]       = f32_to_bf16(q);
    rp[128 + t] = f32_to_bf16(kk);
}

// ---------------------------------------------------------------------------
// 7) correlation volume (5x5 window, l2-normalized, edge pad) -> ch 256..280
// ---------------------------------------------------------------------------
__global__ void k_corr(unsigned short* __restrict__ reg) {
    int gp = blockIdx.x;
    int bb = gp / HWp, p = gp % HWp;
    int y = p / WW, x = p % WW;
    const unsigned short* rp = reg + (size_t)gp * REGP;
    __shared__ float qs[EMB];
    __shared__ float red[32];
    __shared__ float qinv;
    int t = threadIdx.x;                 // 32
    float part = 0.f;
    for (int c = t; c < EMB; c += 32) {
        float v = bf16_to_f32(rp[c]);
        qs[c] = v; part += v * v;
    }
    red[t] = part;
    __syncthreads();
    if (t == 0) {
        float s = 0.f;
        for (int i = 0; i < 32; ++i) s += red[i];
        qinv = 1.f / fmaxf(sqrtf(s), 1e-12f);
    }
    __syncthreads();
    if (t < NOFF) {
        int dy = t / 5 - 2, dx = t % 5 - 2;
        int yy = min(max(y + dy, 0), HH - 1);
        int xx = min(max(x + dx, 0), WW - 1);
        const unsigned short* kp = reg + ((size_t)bb * HWp + yy * WW + xx) * REGP + 128;
        float nk = 0.f, dot = 0.f;
        for (int c = 0; c < EMB; ++c) {
            float kv = bf16_to_f32(kp[c]);
            nk += kv * kv; dot += qs[c] * kv;
        }
        float corr = dot * qinv / fmaxf(sqrtf(nk), 1e-12f);
        reg[(size_t)gp * REGP + 256 + t] = f32_to_bf16(corr);
    }
}

// ---------------------------------------------------------------------------
// 8) weight pre-shuffle into per-lane WMMA B-fragments.
//    Layout: dst[(((tap*nchunk + chunk)*NT + ntile)*32 + lane)*16 + e]
//    (per-tap slices contiguous -> trivially stageable into LDS)
// ---------------------------------------------------------------------------
__global__ void k_prep_w(const float* __restrict__ w, unsigned short* __restrict__ dst,
                         int Cout, int Cin, int nchunk) {
    int NT = Cout / 16;
    int total = NT * 9 * nchunk * 512;
    int idx = blockIdx.x * blockDim.x + threadIdx.x;
    if (idx >= total) return;
    int e     = idx & 15;
    int lane  = (idx >> 4) & 31;
    int rest  = idx >> 9;
    int ntile = rest % NT;   rest /= NT;
    int chunk = rest % nchunk;
    int tap   = rest / nchunk;
    int n = lane & 15, half = lane >> 4;
    int pp = e >> 1, wb = e & 1;
    int koff = (pp < 4) ? (2 * pp + wb + 8 * half) : (16 + 2 * (pp - 4) + wb + 8 * half);
    int cin = chunk * 32 + koff;
    int cout = ntile * 16 + n;
    int ky = tap / 3, kx = tap % 3;
    float v = (cin < Cin) ? w[(((size_t)cout * Cin + cin) * 3 + ky) * 3 + kx] : 0.f;
    dst[idx] = f32_to_bf16(v);
}

// ---------------------------------------------------------------------------
// 9) 3x3 conv (pad=1) as implicit GEMM on v_wmma_f32_16x16x32_bf16.
//    Block = 4 waves x (2 M-tiles of 16 px) = 128 consecutive pixels of a row.
//    Weights for each tap staged into LDS (<=48KB groups), B-fragments read
//    from LDS and reused for 2 WMMAs; next stage prefetched toward the WGP.
// ---------------------------------------------------------------------------
template <int NCHUNK, int NT, bool OUT_BF16>
__global__ void __launch_bounds__(128) k_conv3x3_wmma(
    const unsigned short* __restrict__ in, const unsigned short* __restrict__ wprep,
    const float* __restrict__ bias, void* __restrict__ outp,
    int Cpad, int CoutPad) {

    // chunks staged per LDS fill (keep static shared <= 48 KB)
    constexpr int CSTAGE   = (NCHUNK * NT * 512 * 2 <= 49152) ? NCHUNK
                                                              : (49152 / (NT * 512 * 2));
    constexpr int STAGE_ELE = CSTAGE * NT * 512;        // bf16 elements per fill
    __shared__ __align__(16) unsigned short smem[STAGE_ELE];

    const int tid  = threadIdx.x;
    const int wave = tid >> 5;
    const int lane = tid & 31;
    const int lm = lane & 15, half = lane >> 4;

    // block covers 128 consecutive pixels of one row: blk in [0, B*H*2)
    const int blk  = blockIdx.x;
    const int xseg = blk & 1;
    const int yy   = (blk >> 1) & (HH - 1);
    const int bb   = blk >> 9;
    const int px0  = xseg * 128 + wave * 32 + lm;       // M-tile 0 pixel (this lane)

    v8f acc[2][NT];
#pragma unroll
    for (int i = 0; i < 2; ++i)
#pragma unroll
        for (int n = 0; n < NT; ++n) acc[i][n] = (v8f){0, 0, 0, 0, 0, 0, 0, 0};

    constexpr size_t TOTAL_ELE = (size_t)9 * NCHUNK * NT * 512;

    for (int tap = 0; tap < 9; ++tap) {
        const int sy = yy + tap / 3 - 1;
        const int dx = tap % 3 - 1;
        const bool rowok = (sy >= 0) & (sy < HH);
        int  sx[2];
        bool inb[2];
        const unsigned short* ab[2];
#pragma unroll
        for (int i = 0; i < 2; ++i) {
            sx[i]  = px0 + 16 * i + dx;
            inb[i] = rowok & (sx[i] >= 0) & (sx[i] < WW);
            ab[i]  = in + ((size_t)((bb * HH + sy) * WW) + sx[i]) * Cpad + half * 8;
        }

        for (int cbase = 0; cbase < NCHUNK; cbase += CSTAGE) {
            const int cnum = (NCHUNK - cbase < CSTAGE) ? (NCHUNK - cbase) : CSTAGE;
            const size_t slice0 = ((size_t)tap * NCHUNK + cbase) * NT * 512;
            // ---- stage this weight slice into LDS (all 128 threads) ----
            {
                const uint4* src = (const uint4*)(wprep + slice0);
                uint4* dst = (uint4*)smem;
                const int nv = cnum * NT * 512 / 8;     // uint4 count
                for (int i = tid; i < nv; i += 128) dst[i] = src[i];
                // prefetch next slice toward this WGP while we compute
                const size_t nxt0 = slice0 + (size_t)cnum * NT * 512;
                for (size_t eo = nxt0 + (size_t)tid * 128;
                     eo < nxt0 + (size_t)CSTAGE * NT * 512 && eo < TOTAL_ELE;
                     eo += (size_t)128 * 128)
                    __builtin_prefetch((const void*)(wprep + eo), 0, 1);
            }
            __syncthreads();

            for (int ch = 0; ch < cnum; ++ch) {
                ABfrag A[2];
#pragma unroll
                for (int i = 0; i < 2; ++i) {
                    if (inb[i]) {
                        const unsigned short* ap = ab[i] + (cbase + ch) * 32;
                        A[i].u[0] = *(const uint4*)(ap);
                        A[i].u[1] = *(const uint4*)(ap + 16);
                    } else {
                        A[i].u[0] = make_uint4(0, 0, 0, 0);
                        A[i].u[1] = make_uint4(0, 0, 0, 0);
                    }
                }
#pragma unroll
                for (int nt = 0; nt < NT; ++nt) {
                    ABfrag Bf;
                    const uint4* bp =
                        (const uint4*)(smem + (((size_t)ch * NT + nt) * 32 + lane) * 16);
                    Bf.u[0] = bp[0];
                    Bf.u[1] = bp[1];
                    acc[0][nt] = __builtin_amdgcn_wmma_f32_16x16x32_bf16(
                        false, A[0].v, false, Bf.v, (short)0, acc[0][nt], false, false);
                    acc[1][nt] = __builtin_amdgcn_wmma_f32_16x16x32_bf16(
                        false, A[1].v, false, Bf.v, (short)0, acc[1][nt], false, false);
                }
            }
            __syncthreads();
        }
    }

#pragma unroll
    for (int i = 0; i < 2; ++i) {
#pragma unroll
        for (int nt = 0; nt < NT; ++nt) {
            const int cout = nt * 16 + lm;              // D: lane holds N = lane%16
            const float bv = bias[cout];
#pragma unroll
            for (int j = 0; j < 8; ++j) {               // VGPR j -> M = j + 8*half
                const int ox = xseg * 128 + wave * 32 + 16 * i + j + 8 * half;
                float v = acc[i][nt][j] + bv;
                v = v > 0.f ? v : 0.2f * v;             // leaky_relu(0.2)
                const size_t oidx =
                    ((size_t)((bb * HH + yy) * WW) + ox) * CoutPad + cout;
                if (OUT_BF16) ((unsigned short*)outp)[oidx] = f32_to_bf16(v);
                else          ((float*)outp)[oidx] = v;
            }
        }
    }
}

// ---------------------------------------------------------------------------
// 10) r3 1x1 (64->2) + tanh + scale -> flow (NHWC, 2 per pixel)
// ---------------------------------------------------------------------------
__global__ void k_flow(const float* __restrict__ h2, const float* __restrict__ w3,
                       const float* __restrict__ b3, const float* __restrict__ diff,
                       float* __restrict__ flow) {
    int idx = blockIdx.x * blockDim.x + threadIdx.x;
    if (idx >= BB * HWp) return;
    const float* hp = h2 + (size_t)idx * 64;
    float d = diff[idx];
    for (int c = 0; c < 2; ++c) {
        float s = b3[c];
        for (int i = 0; i < 64; ++i) s += hp[i] * w3[c * 64 + i];
        flow[(size_t)idx * 2 + c] = tanhf(s) * MAXFLOW * d;
    }
}

// ---------------------------------------------------------------------------
// 11) reflect-pad bilinear warp of x31 by flow -> d_out (B,31,H,W) f32
// ---------------------------------------------------------------------------
static __device__ __forceinline__ float reflectc(float coord, int size) {
    float span = (float)size;
    float c = fabsf(coord + 0.5f);
    float extra = fmodf(c, span);
    float flips = floorf(c / span);
    float r = (fmodf(flips, 2.0f) == 0.0f) ? (extra - 0.5f) : (span - extra - 0.5f);
    return clampf(r, 0.0f, (float)size - 1.0f);
}

__global__ void k_warp(const float* __restrict__ x31, const float* __restrict__ flow,
                       float* __restrict__ out) {
    int idx = blockIdx.x * blockDim.x + threadIdx.x;
    if (idx >= BB * HWp) return;
    int p = idx % HWp, bb = idx / HWp;
    int y = p / WW, x = p % WW;
    float fx = flow[(size_t)idx * 2 + 0];
    float fy = flow[(size_t)idx * 2 + 1];
    float ix = reflectc((float)x + fx, WW);
    float iy = reflectc((float)y + fy, HH);
    float x0f = floorf(ix), y0f = floorf(iy);
    float wx = ix - x0f, wy = iy - y0f;
    int x0 = min(max((int)x0f, 0), WW - 1);
    int x1 = min(x0 + 1, WW - 1);
    int y0 = min(max((int)y0f, 0), HH - 1);
    int y1 = min(y0 + 1, HH - 1);
    float w00 = (1 - wy) * (1 - wx), w01 = (1 - wy) * wx;
    float w10 = wy * (1 - wx), w11 = wy * wx;
    for (int c = 0; c < C31; ++c) {
        const float* xb = x31 + ((size_t)bb * C31 + c) * HWp;
        float v = xb[y0 * WW + x0] * w00 + xb[y0 * WW + x1] * w01 +
                  xb[y1 * WW + x0] * w10 + xb[y1 * WW + x1] * w11;
        out[((size_t)bb * C31 + c) * HWp + p] = v;
    }
}

// ---------------------------------------------------------------------------
// launch
// ---------------------------------------------------------------------------
extern "C" void kernel_launch(void* const* d_in, const int* in_sizes, int n_in,
                              void* d_out, int out_size, void* d_ws, size_t ws_size,
                              hipStream_t stream) {
    const float* x   = (const float*)d_in[0];
    const float* y   = (const float*)d_in[1];
    const float* SRF = (const float*)d_in[2];
    const float* chw = (const float*)d_in[3];
    const float* chb = (const float*)d_in[4];
    const float* qw  = (const float*)d_in[5];
    const float* qb  = (const float*)d_in[6];
    const float* kw  = (const float*)d_in[7];
    const float* kb  = (const float*)d_in[8];
    const float* r1w = (const float*)d_in[9];
    const float* r1b = (const float*)d_in[10];
    const float* r2w = (const float*)d_in[11];
    const float* r2b = (const float*)d_in[12];
    const float* r3w = (const float*)d_in[13];
    const float* r3b = (const float*)d_in[14];
    float* out = (float*)d_out;

    char* ws = (char*)d_ws;
    float*          x31   = (float*)(ws + OFF_X31);
    float*          xp    = (float*)(ws + OFF_XP);
    float*          xa    = (float*)(ws + OFF_XA);
    float*          stats = (float*)(ws + OFF_ST);
    float*          ssim3 = (float*)(ws + OFF_SS);
    float*          diff  = (float*)(ws + OFF_DF);
    unsigned short* reg   = (unsigned short*)(ws + OFF_REG);
    unsigned short* h1    = (unsigned short*)(ws + OFF_H1);
    float*          h2    = (float*)(ws + OFF_H2);
    float*          flow  = (float*)(ws + OFF_FL);
    unsigned short* wp1   = (unsigned short*)(ws + OFF_WP1);
    unsigned short* wp2   = (unsigned short*)(ws + OFF_WP2);

    const int TPB = 256;

    // weight pre-shuffle (deterministic, every call)
    {
        int t1 = 8 * 9 * 9 * 512;
        k_prep_w<<<(t1 + TPB - 1) / TPB, TPB, 0, stream>>>(r1w, wp1, 128, REG, 9);
        int t2 = 4 * 9 * 4 * 512;
        k_prep_w<<<(t2 + TPB - 1) / TPB, TPB, 0, stream>>>(r2w, wp2, 64, 128, 4);
    }

    // front end
    k_conv_half<<<(BB * C31 * HWp + TPB - 1) / TPB, TPB, 0, stream>>>(x, chw, chb, x31);
    k_project<<<(BB * CM * HWp + TPB - 1) / TPB, TPB, 0, stream>>>(x31, SRF, xp);
    k_stats<<<BB * CM, 256, 0, stream>>>(xp, y, stats);
    k_align<<<(BB * CM * HWp + TPB - 1) / TPB, TPB, 0, stream>>>(xp, stats, xa);
    k_ssim<<<(BB * CM * HWp + TPB - 1) / TPB, TPB, 0, stream>>>(xa, y, ssim3);
    k_diff<<<(BB * HWp + TPB - 1) / TPB, TPB, 0, stream>>>(ssim3, diff, reg);

    // reg_in assembly
    k_qk<<<BB * HWp, 128, 0, stream>>>(x31, y, qw, qb, kw, kb, reg);
    k_corr<<<BB * HWp, 32, 0, stream>>>(reg);

    // heavy 3x3 convs on WMMA: 128-pixel blocks (4 waves x 2 M-tiles)
    const int nblocks = BB * HH * 2;    // 2048
    k_conv3x3_wmma<9, 8, true><<<nblocks, 128, 0, stream>>>(reg, wp1, r1b, (void*)h1, REGP, 128);
    k_conv3x3_wmma<4, 4, false><<<nblocks, 128, 0, stream>>>(h1, wp2, r2b, (void*)h2, 128, 64);

    // flow head + warp
    k_flow<<<(BB * HWp + TPB - 1) / TPB, TPB, 0, stream>>>(h2, r3w, r3b, diff, flow);
    k_warp<<<(BB * HWp + TPB - 1) / TPB, TPB, 0, stream>>>(x31, flow, out);

    (void)in_sizes; (void)n_in; (void)out_size; (void)ws_size;
}
